// MoEBlock_11579231830574
// MI455X (gfx1250) — compile-verified, matrix operand order checked by reading, not run on hardware
//
#include <hip/hip_runtime.h>
#include <math.h>

// ---------------- problem constants (match reference) ----------------
#define HID    2048
#define NHEADS 16
#define HDIM   128
#define NEXP   8
#define INTER  4096
#define SEQ    2048
#define NBATCH 2
#define NTOK   (NBATCH * SEQ)
#define EPSV   1e-5f

// ---------------- vector types ----------------
typedef __attribute__((ext_vector_type(16))) __bf16 v16bf;
typedef __attribute__((ext_vector_type(8)))  float  v8f;
typedef __attribute__((ext_vector_type(4)))  float  f32x4;
typedef __attribute__((ext_vector_type(4)))  unsigned int u32x4;
typedef __attribute__((ext_vector_type(2)))  unsigned int u32x2;
typedef __attribute__((ext_vector_type(4)))  unsigned int v4u;
typedef __attribute__((ext_vector_type(8)))  int v8i;
typedef __attribute__((ext_vector_type(4)))  int v4i;

union FragBF { u32x4 q[2]; v16bf v; };
union H8 { u32x4 q; unsigned short s[8]; };

__device__ __forceinline__ unsigned short f2bf(float f) {
    unsigned int u = __float_as_uint(f);
    u += 0x7FFFu + ((u >> 16) & 1u);          // round-to-nearest-even
    return (unsigned short)(u >> 16);
}
__device__ __forceinline__ unsigned int pack2(float a, float b) {
    return (unsigned int)f2bf(a) | ((unsigned int)f2bf(b) << 16);
}
__device__ __forceinline__ float bf2f(unsigned short h) {
    return __uint_as_float(((unsigned int)h) << 16);
}

// 16x32 bf16 WMMA operand from LDS (row-major, rowStride in halfwords).
// Lane l: row (l&15); VGPR0-3 = K (l/16)*8..+7, VGPR4-7 = K 16+(l/16)*8..+7.
__device__ __forceinline__ v16bf load_frag(const unsigned short* p, int lane, int rowStride) {
    int r  = lane & 15;
    int ko = (lane >> 4) << 3;
    FragBF f;
    f.q[0] = *(const u32x4*)(p + r * rowStride + ko);
    f.q[1] = *(const u32x4*)(p + r * rowStride + 16 + ko);
    return f.v;
}

// ---------------------------------------------------------------------
// Tensor Data Mover: DMA a 2D bf16 tile [128 rows x 32 cols], row stride
// strideElems, from global into LDS at byte offset ldsoff (row-major,
// 32 halfwords per row, rows contiguous). D# packing per ISA 08 §8.
// This toolchain exposes the 6-arg builtin:
//   (uint32x4 g0, int32x8 g1, int32x4, int32x4, int32x8, i32 cpol)
// ---------------------------------------------------------------------
__device__ __forceinline__ void tdm_load_tile(unsigned ldsoff,
                                              const unsigned short* gsrc,
                                              int strideElems) {
    unsigned long long ga = (unsigned long long)(size_t)gsrc;
    v4u g0;
    g0.x = 1u;                                        // count=1 (valid user D#)
    g0.y = ldsoff;                                    // lds_addr (bytes)
    g0.z = (unsigned)ga;                              // global_addr[31:0]
    g0.w = (unsigned)((ga >> 32) & 0x1FFFFFFull) | (2u << 30);  // addr[56:32] | type=2
    v8i g1;
    g1[0] = (1 << 16);                                // data_size=1 (2 bytes)
    g1[1] = (int)(((unsigned)strideElems & 0xFFFFu) << 16);     // tensor_dim0 lo16
    g1[2] = (int)(((unsigned)strideElems >> 16) & 0xFFFFu);     // tensor_dim0 hi16, td1 lo16=0
    g1[3] = (int)(16u | (32u << 16));                 // tensor_dim1=1<<20 (hi16=16), tile_dim0=32
    g1[4] = 128;                                      // tile_dim1=128, tile_dim2=0
    g1[5] = strideElems;                              // tensor_dim0_stride lo32
    g1[6] = 0;
    g1[7] = 0;
    v4i z4; z4[0] = 0; z4[1] = 0; z4[2] = 0; z4[3] = 0;   // groups 2/3: unused (2D tile)
    v8i z8;
    #pragma unroll
    for (int i = 0; i < 8; ++i) z8[i] = 0;
    __builtin_amdgcn_tensor_load_to_lds(g0, g1, z4, z4, z8, 0);
}
__device__ __forceinline__ unsigned lds_addr_of(const void* p) {
    return (unsigned)(size_t)p;   // low 32 bits of flat LDS address = LDS byte offset
}

// =====================================================================
// Generic NT GEMM (bf16 in, WMMA, f32 accum), TDM-staged + double buffer
//   C[M,N] = A[M,K] * W[N,K]^T
//   mode 0: C(f32) = acc
//   mode 1: C(f32) = aux[idx] + acc
//   mode 2: C(f32) += aux[row*NEXP] * acc
//   mode 3: C(bf16) = acc
// Requires M%128==0, N%128==0, K%32==0.
// =====================================================================
#define BM 128
#define BN 128
#define BK 32

__global__ __launch_bounds__(256)
void gemm_bf16_nt(const unsigned short* __restrict__ A,
                  const unsigned short* __restrict__ W,
                  void* __restrict__ Cout, const float* __restrict__ aux,
                  int M, int N, int K, int mode)
{
    __shared__ __align__(16) unsigned short sA[2][BM * BK];
    __shared__ __align__(16) unsigned short sB[2][BN * BK];

    const int tid  = threadIdx.x;
    const int lane = tid & 31;
    const int wave = tid >> 5;
    const int wm   = wave >> 2;      // 0..1  (64 rows)
    const int wn   = wave & 3;       // 0..3  (32 cols)
    const int m0   = blockIdx.y * BM;
    const int n0   = blockIdx.x * BN;

    v8f acc[4][2];
    #pragma unroll
    for (int i = 0; i < 4; ++i)
        #pragma unroll
        for (int j = 0; j < 2; ++j)
            #pragma unroll
            for (int v = 0; v < 8; ++v) acc[i][j][v] = 0.0f;

    const unsigned short* Abase = A + (size_t)m0 * (size_t)K;
    const unsigned short* Wbase = W + (size_t)n0 * (size_t)K;
    const int nk = K / BK;

    if (wave == 0) {                 // prologue: fill buffer 0
        tdm_load_tile(lds_addr_of(&sA[0][0]), Abase, K);
        tdm_load_tile(lds_addr_of(&sB[0][0]), Wbase, K);
    }

    for (int step = 0; step < nk; ++step) {
        const int buf = step & 1;
        if (wave == 0) {
            if (step + 1 < nk) {     // prefetch next tile pair into other buffer
                tdm_load_tile(lds_addr_of(&sA[buf ^ 1][0]), Abase + (size_t)(step + 1) * BK, K);
                tdm_load_tile(lds_addr_of(&sB[buf ^ 1][0]), Wbase + (size_t)(step + 1) * BK, K);
                __builtin_amdgcn_s_wait_tensorcnt(2);   // current buffer's pair done
            } else {
                __builtin_amdgcn_s_wait_tensorcnt(0);
            }
        }
        __syncthreads();             // publish TDM-written LDS to all waves

        v16bf af[4], bfv[2];
        #pragma unroll
        for (int i = 0; i < 4; ++i)
            af[i] = load_frag(&sA[buf][(wm * 64 + i * 16) * BK], lane, BK);
        #pragma unroll
        for (int j = 0; j < 2; ++j)
            bfv[j] = load_frag(&sB[buf][(wn * 32 + j * 16) * BK], lane, BK);

        #pragma unroll
        for (int i = 0; i < 4; ++i)
            #pragma unroll
            for (int j = 0; j < 2; ++j)
                acc[i][j] = __builtin_amdgcn_wmma_f32_16x16x32_bf16(
                    false, af[i], false, bfv[j], (short)0, acc[i][j], false, false);
        __syncthreads();             // reads done before buf^1 is overwritten
    }

    const int lr = (lane >> 4) << 3;
    const int lc = lane & 15;
    float* Cf = (float*)Cout;
    unsigned short* Ch = (unsigned short*)Cout;
    #pragma unroll
    for (int i = 0; i < 4; ++i) {
        #pragma unroll
        for (int j = 0; j < 2; ++j) {
            #pragma unroll
            for (int v = 0; v < 8; ++v) {
                int row = m0 + wm * 64 + i * 16 + lr + v;
                int col = n0 + wn * 32 + j * 16 + lc;
                size_t idx = (size_t)row * (size_t)N + col;
                float val = acc[i][j][v];
                if (mode == 1)      Cf[idx] = aux[idx] + val;
                else if (mode == 2) Cf[idx] += aux[(size_t)row * NEXP] * val;
                else if (mode == 3) Ch[idx] = f2bf(val);
                else                Cf[idx] = val;
            }
        }
    }
}

// =====================================================================
// RMSNorm: f32 in -> bf16 out, one block per token
// =====================================================================
__global__ __launch_bounds__(256)
void rmsnorm_kernel(const float* __restrict__ x, const float* __restrict__ w,
                    unsigned short* __restrict__ y)
{
    int t = blockIdx.x;
    const float* xr = x + (size_t)t * HID;
    float ss = 0.f;
    for (int i = threadIdx.x; i < HID; i += 256) { float v = xr[i]; ss += v * v; }
    #pragma unroll
    for (int m = 16; m >= 1; m >>= 1) ss += __shfl_xor(ss, m, 32);
    __shared__ float red[8];
    if ((threadIdx.x & 31) == 0) red[threadIdx.x >> 5] = ss;
    __syncthreads();
    float tot = 0.f;
    #pragma unroll
    for (int i = 0; i < 8; ++i) tot += red[i];
    float rs = rsqrtf(tot * (1.0f / HID) + EPSV);
    unsigned short* yr = y + (size_t)t * HID;
    for (int i = threadIdx.x; i < HID; i += 256) yr[i] = f2bf(xr[i] * rs * w[i]);
}

// =====================================================================
// f32 -> bf16 convert (weights), n must be a multiple of 4
// =====================================================================
__global__ void cvt_f32_bf16_kernel(const float* __restrict__ s,
                                    unsigned short* __restrict__ d, size_t n4)
{
    size_t i = (size_t)blockIdx.x * blockDim.x + threadIdx.x;
    size_t stride = (size_t)gridDim.x * blockDim.x;
    for (; i < n4; i += stride) {
        f32x4 v = *(const f32x4*)(s + i * 4);
        u32x2 t; t.x = pack2(v.x, v.y); t.y = pack2(v.z, v.w);
        *(u32x2*)(d + i * 4) = t;
    }
}

// =====================================================================
// Flash-style causal attention (bf16 q/k/v in, bf16 h out)
// one block = (64 q-rows, 1 head, 1 batch); 4 waves x 16 q-rows
// =====================================================================
#define BQ  64
#define BKV 32
#define NEG_BIG (-3.0e38f)

__global__ __launch_bounds__(128)
void flash_attn_kernel(const unsigned short* __restrict__ Q,
                       const unsigned short* __restrict__ Kx,
                       const unsigned short* __restrict__ V,
                       unsigned short* __restrict__ O)
{
    __shared__ __align__(16) unsigned short sQ [BQ  * HDIM];   // 16 KB
    __shared__ __align__(16) unsigned short sK [BKV * HDIM];   //  8 KB
    __shared__ __align__(16) unsigned short sVt[HDIM * BKV];   //  8 KB (transposed)
    __shared__ __align__(16) unsigned short sP [4 * 16 * BKV]; //  4 KB

    const int tid  = threadIdx.x;
    const int lane = tid & 31;
    const int wave = tid >> 5;
    const int q0   = blockIdx.x * BQ;
    const int h    = blockIdx.y;
    const int b    = blockIdx.z;

    const unsigned short* Qb = Q + (size_t)b * SEQ * HID + (size_t)h * HDIM;
    const unsigned short* Kb = Kx + (size_t)b * SEQ * HID + (size_t)h * HDIM;
    const unsigned short* Vb = V + (size_t)b * SEQ * HID + (size_t)h * HDIM;

    // stage Q tile (64x128 halfs): straight b128 copies
    #pragma unroll
    for (int p = 0; p < 8; ++p) {
        int idx = p * 128 + tid;
        int r = idx >> 4, cc = (idx & 15) * 8;
        *(u32x4*)(sQ + r * HDIM + cc) = *(const u32x4*)(Qb + (size_t)(q0 + r) * HID + cc);
    }
    __syncthreads();

    v8f Oacc[8];
    float mI[8], lI[8];
    #pragma unroll
    for (int nt = 0; nt < 8; ++nt)
        #pragma unroll
        for (int v = 0; v < 8; ++v) Oacc[nt][v] = 0.f;
    #pragma unroll
    for (int v = 0; v < 8; ++v) { mI[v] = NEG_BIG; lI[v] = 0.f; }

    const int lr = (lane >> 4) << 3;
    const int lc = lane & 15;
    const float scale = 0.08838834764831845f;   // 1/sqrt(128)
    const int nkt = (q0 + BQ) / BKV;

    for (int kt = 0; kt < nkt; ++kt) {
        const int k0 = kt * BKV;
        // stage K (32x128) copy + V^T (128x32) scatter
        #pragma unroll
        for (int p = 0; p < 4; ++p) {
            int idx = p * 128 + tid;
            int r = idx >> 4, cc = (idx & 15) * 8;
            *(u32x4*)(sK + r * HDIM + cc) =
                *(const u32x4*)(Kb + (size_t)(k0 + r) * HID + cc);
            H8 vv; vv.q = *(const u32x4*)(Vb + (size_t)(k0 + r) * HID + cc);
            #pragma unroll
            for (int j = 0; j < 8; ++j) sVt[(cc + j) * BKV + r] = vv.s[j];
        }
        __syncthreads();

        // S = Q * K^T (16 q-rows x 32 keys per wave), head-dim K=128 -> 4 chained WMMAs
        v8f s[2];
        #pragma unroll
        for (int nt = 0; nt < 2; ++nt)
            #pragma unroll
            for (int v = 0; v < 8; ++v) s[nt][v] = 0.f;
        #pragma unroll
        for (int kc = 0; kc < 4; ++kc) {
            v16bf aq = load_frag(sQ + (wave * 16) * HDIM + kc * 32, lane, HDIM);
            #pragma unroll
            for (int nt = 0; nt < 2; ++nt) {
                v16bf bk = load_frag(sK + (nt * 16) * HDIM + kc * 32, lane, HDIM);
                s[nt] = __builtin_amdgcn_wmma_f32_16x16x32_bf16(
                    false, aq, false, bk, (short)0, s[nt], false, false);
            }
        }

        // online softmax, elementwise in the C layout (row = v + 8*(lane/16))
        #pragma unroll
        for (int v = 0; v < 8; ++v) {
            int qrow = q0 + wave * 16 + lr + v;
            float s0 = s[0][v] * scale;
            float s1 = s[1][v] * scale;
            if (k0 + lc      > qrow) s0 = NEG_BIG;
            if (k0 + 16 + lc > qrow) s1 = NEG_BIG;
            float mx = fmaxf(s0, s1);
            #pragma unroll
            for (int msk = 8; msk >= 1; msk >>= 1)
                mx = fmaxf(mx, __shfl_xor(mx, msk, 32));
            float mNew = fmaxf(mI[v], mx);
            float fac  = __expf(mI[v] - mNew);
            float p0   = __expf(s0 - mNew);
            float p1   = __expf(s1 - mNew);
            float rs   = p0 + p1;
            #pragma unroll
            for (int msk = 8; msk >= 1; msk >>= 1)
                rs += __shfl_xor(rs, msk, 32);
            lI[v] = lI[v] * fac + rs;
            mI[v] = mNew;
            #pragma unroll
            for (int nt = 0; nt < 8; ++nt) Oacc[nt][v] *= fac;
            int prow = lr + v;
            sP[wave * 16 * BKV + prow * BKV + lc]      = f2bf(p0);
            sP[wave * 16 * BKV + prow * BKV + 16 + lc] = f2bf(p1);
        }
        __syncthreads();

        // O += P * V
        v16bf ap = load_frag(sP + wave * 16 * BKV, lane, BKV);
        #pragma unroll
        for (int nt = 0; nt < 8; ++nt) {
            v16bf bv = load_frag(sVt + (nt * 16) * BKV, lane, BKV);
            Oacc[nt] = __builtin_amdgcn_wmma_f32_16x16x32_bf16(
                false, ap, false, bv, (short)0, Oacc[nt], false, false);
        }
        __syncthreads();
    }

    unsigned short* Ob = O + (size_t)b * SEQ * HID + (size_t)h * HDIM;
    #pragma unroll
    for (int nt = 0; nt < 8; ++nt) {
        #pragma unroll
        for (int v = 0; v < 8; ++v) {
            int qr = q0 + wave * 16 + lr + v;
            int dc = nt * 16 + lc;
            Ob[(size_t)qr * HID + dc] = f2bf(Oacc[nt][v] / lI[v]);
        }
    }
}

// =====================================================================
// Router: logits = xn @ Wr^T (Wr f32, xn bf16), top-2 softmax -> cw
// =====================================================================
__global__ __launch_bounds__(256)
void router_kernel(const unsigned short* __restrict__ xn, const float* __restrict__ Wr,
                   float* __restrict__ cw)
{
    int t = blockIdx.x;
    const unsigned short* xr = xn + (size_t)t * HID;
    float part[NEXP];
    #pragma unroll
    for (int e = 0; e < NEXP; ++e) part[e] = 0.f;
    for (int i = threadIdx.x; i < HID; i += 256) {
        float xv = bf2f(xr[i]);
        #pragma unroll
        for (int e = 0; e < NEXP; ++e) part[e] += xv * Wr[(size_t)e * HID + i];
    }
    __shared__ float red[8][NEXP];
    #pragma unroll
    for (int e = 0; e < NEXP; ++e) {
        float v = part[e];
        #pragma unroll
        for (int m = 16; m >= 1; m >>= 1) v += __shfl_xor(v, m, 32);
        part[e] = v;
    }
    if ((threadIdx.x & 31) == 0)
        for (int e = 0; e < NEXP; ++e) red[threadIdx.x >> 5][e] = part[e];
    __syncthreads();
    if (threadIdx.x == 0) {
        float lg[NEXP];
        for (int e = 0; e < NEXP; ++e) {
            float s = 0.f;
            for (int w = 0; w < 8; ++w) s += red[w][e];
            lg[e] = s;
        }
        int i0 = 0;
        for (int e = 1; e < NEXP; ++e) if (lg[e] > lg[i0]) i0 = e;
        int i1 = -1;
        for (int e = 0; e < NEXP; ++e) {
            if (e == i0) continue;
            if (i1 < 0 || lg[e] > lg[i1]) i1 = e;
        }
        float p1 = __expf(lg[i1] - lg[i0]);
        float inv = 1.0f / (1.0f + p1);
        for (int e = 0; e < NEXP; ++e) cw[(size_t)t * NEXP + e] = 0.f;
        cw[(size_t)t * NEXP + i0] = inv;
        cw[(size_t)t * NEXP + i1] = p1 * inv;
    }
}

// =====================================================================
// SwiGLU: out_bf16 = silu(hg_f32) * hu_f32
// =====================================================================
__global__ void swiglu_kernel(const float* __restrict__ hg, const float* __restrict__ hu,
                              unsigned short* __restrict__ outb, size_t n)
{
    size_t i = (size_t)blockIdx.x * blockDim.x + threadIdx.x;
    size_t stride = (size_t)gridDim.x * blockDim.x;
    for (; i < n; i += stride) {
        float g = hg[i];
        outb[i] = f2bf((g / (1.f + __expf(-g))) * hu[i]);
    }
}

// =====================================================================
// Host-side orchestration
// =====================================================================
extern "C" void kernel_launch(void* const* d_in, const int* in_sizes, int n_in,
                              void* d_out, int out_size, void* d_ws, size_t ws_size,
                              hipStream_t stream)
{
    (void)in_sizes; (void)n_in; (void)out_size; (void)ws_size;
    const float* x   = (const float*)d_in[0];
    const float* Wq  = (const float*)d_in[1];
    const float* Wk  = (const float*)d_in[2];
    const float* Wv  = (const float*)d_in[3];
    const float* Wo  = (const float*)d_in[4];
    const float* wl1 = (const float*)d_in[5];
    const float* wl2 = (const float*)d_in[6];
    const float* Wr  = (const float*)d_in[7];
    const float* Wg  = (const float*)d_in[8];
    const float* Wu  = (const float*)d_in[9];
    const float* Wd  = (const float*)d_in[10];
    float* out = (float*)d_out;

    // ---- workspace layout ----
    char* ws = (char*)d_ws;
    const size_t szXnb = (size_t)NTOK * HID * 2;       // xn bf16
    const size_t szR1  = (size_t)NTOK * INTER * 4;     // q/k/v/h bf16  OR  hg f32
    const size_t szHu  = (size_t)NTOK * INTER * 4;     // hu f32
    const size_t szHgb = (size_t)NTOK * INTER * 2;     // swiglu out bf16
    const size_t szWb  = (size_t)INTER * HID * 2;      // bf16 weight staging

    unsigned short* xnb = (unsigned short*)ws;
    char* r1 = ws + szXnb;
    unsigned short* qb = (unsigned short*)r1;
    unsigned short* kb = qb + (size_t)NTOK * HID;
    unsigned short* vb = kb + (size_t)NTOK * HID;
    unsigned short* hb = vb + (size_t)NTOK * HID;
    float* hg  = (float*)r1;                           // reuses q/k/v/h after attention
    float* hu  = (float*)(r1 + szR1);
    unsigned short* hgb  = (unsigned short*)(r1 + szR1 + szHu);
    unsigned short* wbuf = (unsigned short*)(r1 + szR1 + szHu + szHgb);
    float* cw = (float*)(r1 + szR1 + szHu + szHgb + szWb);

    const dim3 blk(256);
    const dim3 gHID(HID / BN, NTOK / BM);
    const dim3 gINT(INTER / BN, NTOK / BM);
    const size_t nW_h = (size_t)HID * HID / 4;         // f32x4 chunks
    const size_t nW_i = (size_t)INTER * HID / 4;

    // 1) xn = rmsnorm(x, w_ln1)  (bf16 out)
    rmsnorm_kernel<<<NTOK, blk, 0, stream>>>(x, wl1, xnb);

    // 2) q,k,v projections (bf16 out), weights converted on the fly
    cvt_f32_bf16_kernel<<<4096, blk, 0, stream>>>(Wq, wbuf, nW_h);
    gemm_bf16_nt<<<gHID, blk, 0, stream>>>(xnb, wbuf, qb, nullptr, NTOK, HID, HID, 3);
    cvt_f32_bf16_kernel<<<4096, blk, 0, stream>>>(Wk, wbuf, nW_h);
    gemm_bf16_nt<<<gHID, blk, 0, stream>>>(xnb, wbuf, kb, nullptr, NTOK, HID, HID, 3);
    cvt_f32_bf16_kernel<<<4096, blk, 0, stream>>>(Wv, wbuf, nW_h);
    gemm_bf16_nt<<<gHID, blk, 0, stream>>>(xnb, wbuf, vb, nullptr, NTOK, HID, HID, 3);

    // 3) causal attention -> hb (bf16)
    dim3 ga(SEQ / BQ, NHEADS, NBATCH);
    flash_attn_kernel<<<ga, dim3(128), 0, stream>>>(qb, kb, vb, hb);

    // 4) out = x + hb @ Wo^T   (f32 out)
    cvt_f32_bf16_kernel<<<4096, blk, 0, stream>>>(Wo, wbuf, nW_h);
    gemm_bf16_nt<<<gHID, blk, 0, stream>>>(hb, wbuf, out, x, NTOK, HID, HID, 1);

    // 5) xn = rmsnorm(out, w_ln2)
    rmsnorm_kernel<<<NTOK, blk, 0, stream>>>(out, wl2, xnb);

    // 6) router combine weights
    router_kernel<<<NTOK, blk, 0, stream>>>(xnb, Wr, cw);

    // 7) dense per-expert SwiGLU, cw-weighted accumulate into out
    for (int e = 0; e < NEXP; ++e) {
        const float* Wge = Wg + (size_t)e * INTER * HID;
        const float* Wue = Wu + (size_t)e * INTER * HID;
        const float* Wde = Wd + (size_t)e * HID * INTER;
        cvt_f32_bf16_kernel<<<4096, blk, 0, stream>>>(Wge, wbuf, nW_i);
        gemm_bf16_nt<<<gINT, blk, 0, stream>>>(xnb, wbuf, hg, nullptr, NTOK, INTER, HID, 0);
        cvt_f32_bf16_kernel<<<4096, blk, 0, stream>>>(Wue, wbuf, nW_i);
        gemm_bf16_nt<<<gINT, blk, 0, stream>>>(xnb, wbuf, hu, nullptr, NTOK, INTER, HID, 0);
        swiglu_kernel<<<16384, blk, 0, stream>>>(hg, hu, hgb, (size_t)NTOK * INTER);
        cvt_f32_bf16_kernel<<<4096, blk, 0, stream>>>(Wde, wbuf, nW_i);
        gemm_bf16_nt<<<gHID, blk, 0, stream>>>(hgb, wbuf, out, cw + e, NTOK, HID, INTER, 2);
    }
}
